// FpgnnModel_6743098654881
// MI455X (gfx1250) — compile-verified, matrix operand order checked by reading, not run on hardware
//
#include <hip/hip_runtime.h>

typedef __attribute__((ext_vector_type(16))) __bf16 v16bf;
typedef __attribute__((ext_vector_type(8)))  __bf16 v8bf;
typedef __attribute__((ext_vector_type(8)))  float  v8f;

union BfFrag { v16bf v; v8bf h[2]; };

// Expand (base_w, spline_w, scaler) -> bf16 W[n][e], e = it*288 + s*32 + f_local
// Rows padded to Npad (zero), e padded to KErow (zero).
__global__ __launch_bounds__(256) void expand_w(
    const float* __restrict__ base_w, const float* __restrict__ spline_w,
    const float* __restrict__ scaler, __bf16* __restrict__ Wx,
    int N, int KIN, int KErow, long total) {
  long t = (long)blockIdx.x * blockDim.x + threadIdx.x;
  if (t >= total) return;
  int n  = (int)(t / KErow);
  int eg = (int)(t % KErow);
  int it = eg / 288;
  int r  = eg % 288;
  int s  = r >> 5;      // slot 0..8 (0 = base, 1..8 = spline)
  int fl = r & 31;
  int f  = it * 32 + fl;
  float val = 0.0f;
  if (n < N && f < KIN) {
    size_t idx = (size_t)n * KIN + f;
    val = (s == 0) ? base_w[idx] : spline_w[idx * 8 + (s - 1)] * scaler[idx];
  }
  Wx[t] = (__bf16)val;
}

// Fused KAN-linear GEMM: Out[M x N] = expand9(X[M x KIN]) @ Wexp^T, optional ReLU.
// Block: 256 threads (8 waves), 64 rows x (NT*32) cols output tile.
// Each wave owns one 16-row tile strip and NT 16-col tiles.
template <int KIN, int NT, bool RELU>
__global__ __launch_bounds__(256) void kan_gemm(
    const float* __restrict__ X, const __bf16* __restrict__ W,
    float* __restrict__ Out, int N) {
  constexpr int NITER   = (KIN + 31) / 32;
  constexpr int KEROW   = NITER * 288;
  constexpr int ASTRIDE = 296;                    // 288 + pad, 16B-aligned rows
  constexpr int BN      = NT * 32;                // 2 wave-groups of NT tiles
  __shared__ __bf16 Alds[64 * ASTRIDE];           // 37,888 bytes

  const int tid  = threadIdx.x;
  const int lane = tid & 31;
  const int wave = tid >> 5;
  const int bm   = blockIdx.y * 64;
  const int bn   = blockIdx.x * BN;

  const int tile_m = wave & 3;                    // 4 row-tiles of 16
  const int nbase  = bn + (wave >> 2) * (NT * 16);

  v8f acc[NT];
#pragma unroll
  for (int t = 0; t < NT; ++t)
#pragma unroll
    for (int e = 0; e < 8; ++e) acc[t][e] = 0.0f;

  for (int it = 0; it < NITER; ++it) {
    const int f0 = it * 32;
    // ---- stage expanded A tile: 64 rows x 32 features x 9 slots ----
    // Uniform cubic B-spline: exactly 4 nonzero bases at any x; scatter them.
#pragma unroll
    for (int i = 0; i < 8; ++i) {
      int p    = tid + i * 256;
      int row  = p >> 5;
      int feat = p & 31;
      int f    = f0 + feat;
      __bf16* ap = &Alds[row * ASTRIDE + feat];   // slot s at ap[s*32]

      float x = 0.0f;
      bool valid = (f < KIN);
      if (valid) {
        const float* xp = &X[(size_t)(bm + row) * KIN + f];
        x = *xp;
        if (it + 1 < NITER) __builtin_prefetch(xp + 32, 0, 0);
      }
      // zero spline slots (overwritten below for the 4 active ones)
#pragma unroll
      for (int s = 1; s < 9; ++s) ap[s * 32] = (__bf16)0.0f;
      // slot 0: silu(x)
      float sl = valid ? (x / (1.0f + __expf(-x))) : 0.0f;
      ap[0] = (__bf16)sl;
      // cubic B-spline window: grid g[j] = (j-3)*0.4 - 1, knots j=0..11,
      // order-3 bases j=0..7.  x in [g_i, g_i+1) -> nonzero j = i-3..i.
      if (valid && x >= -2.2f && x < 2.2f) {
        float t  = (x + 1.0f) * 2.5f;             // (x - g0')/h with 1/h = 2.5
        float fi = __builtin_floorf(t);
        int   iv = (int)fi + 3;                   // interval index, 0..10
        float u  = t - fi;
        float om = 1.0f - u;
        float u2 = u * u, u3 = u2 * u;
        float w0 = om * om * om * (1.0f / 6.0f);
        float w1 = (3.0f * u3 - 6.0f * u2 + 4.0f) * (1.0f / 6.0f);
        float w2 = (-3.0f * u3 + 3.0f * u2 + 3.0f * u + 1.0f) * (1.0f / 6.0f);
        float w3 = u3 * (1.0f / 6.0f);
        float w[4] = {w0, w1, w2, w3};
#pragma unroll
        for (int k = 0; k < 4; ++k) {
          int j = iv - 3 + k;
          if (j >= 0 && j <= 7) ap[(j + 1) * 32] = (__bf16)w[k];
        }
      }
    }
    __syncthreads();

    // ---- 9 WMMA k-steps over the 288 expanded columns ----
    const int arow = (tile_m * 16 + (lane & 15)) * ASTRIDE;
    const int koff = (lane >> 4) * 8;   // lanes 0-15: K 0-7/16-23; 16-31: 8-15/24-31
#pragma unroll
    for (int s = 0; s < 9; ++s) {
      BfFrag a;
      a.h[0] = *reinterpret_cast<const v8bf*>(&Alds[arow + s * 32 + koff]);
      a.h[1] = *reinterpret_cast<const v8bf*>(&Alds[arow + s * 32 + koff + 16]);
#pragma unroll
      for (int t = 0; t < NT; ++t) {
        int n = nbase + t * 16 + (lane & 15);
        const __bf16* wp =
            W + (size_t)n * KEROW + (size_t)it * 288 + s * 32 + koff;
        BfFrag b;
        b.h[0] = *reinterpret_cast<const v8bf*>(wp);
        b.h[1] = *reinterpret_cast<const v8bf*>(wp + 16);
        acc[t] = __builtin_amdgcn_wmma_f32_16x16x32_bf16(
            false, a.v, false, b.v, (short)0, acc[t], false, false);
      }
    }
    __syncthreads();
  }

  // ---- epilogue: C/D layout — lane<16: M=r, lane>=16: M=r+8 ----
#pragma unroll
  for (int t = 0; t < NT; ++t) {
    int n = nbase + t * 16 + (lane & 15);
    if (n < N) {
#pragma unroll
      for (int r = 0; r < 8; ++r) {
        int m = bm + tile_m * 16 + ((lane < 16) ? r : (r + 8));
        float val = acc[t][r];
        if (RELU) val = fmaxf(val, 0.0f);
        Out[(size_t)m * N + n] = val;
      }
    }
  }
}

extern "C" void kernel_launch(void* const* d_in, const int* in_sizes, int n_in,
                              void* d_out, int out_size, void* d_ws, size_t ws_size,
                              hipStream_t stream) {
  (void)in_sizes; (void)n_in; (void)out_size; (void)ws_size;
  const float* fp        = (const float*)d_in[0];
  const float* base_w1   = (const float*)d_in[1];
  const float* spline_w1 = (const float*)d_in[2];
  const float* scaler1   = (const float*)d_in[3];
  const float* base_w2   = (const float*)d_in[4];
  const float* spline_w2 = (const float*)d_in[5];
  const float* scaler2   = (const float*)d_in[6];

  constexpr int BATCH = 8192, K1 = 2513, N1 = 512, K2 = 512, N2 = 300;
  constexpr int KE1 = ((K1 + 31) / 32) * 288;   // 22752
  constexpr int KE2 = ((K2 + 31) / 32) * 288;   // 4608
  constexpr int NP1 = 512;                      // padded to BN=256
  constexpr int NP2 = 384;                      // padded to BN=128

  char* ws = (char*)d_ws;
  __bf16* W1x = (__bf16*)ws;
  size_t  off = (size_t)NP1 * KE1 * sizeof(__bf16);   // 23,298,048 B (16B-aligned)
  __bf16* W2x = (__bf16*)(ws + off);
  off += (size_t)NP2 * KE2 * sizeof(__bf16);          // + 3,538,944 B
  float*  H   = (float*)(ws + off);                   // 8192*512 f32

  {
    long tot = (long)NP1 * KE1;
    expand_w<<<dim3((unsigned)((tot + 255) / 256)), 256, 0, stream>>>(
        base_w1, spline_w1, scaler1, W1x, N1, K1, KE1, tot);
  }
  {
    long tot = (long)NP2 * KE2;
    expand_w<<<dim3((unsigned)((tot + 255) / 256)), 256, 0, stream>>>(
        base_w2, spline_w2, scaler2, W2x, N2, K2, KE2, tot);
  }
  // Layer 1: BN=256 (8 n-tiles/wave) -> expansion staged only 2x per row strip.
  kan_gemm<K1, 8, true ><<<dim3(NP1 / 256, BATCH / 64), 256, 0, stream>>>(fp, W1x, H, N1);
  // Layer 2: BN=128 (4 n-tiles/wave), N=300 padded to 384.
  kan_gemm<K2, 4, false><<<dim3(NP2 / 128, BATCH / 64), 256, 0, stream>>>(H, W2x, (float*)d_out, N2);
}
